// GCNEncoder_11862699671809
// MI455X (gfx1250) — matrix operand order repaired and checked by reading, not compiled, not run on hardware
//
#include <hip/hip_runtime.h>
#include <hip/hip_bf16.h>
#include <stdint.h>

typedef __attribute__((ext_vector_type(16))) __bf16 v16bf;
typedef __attribute__((ext_vector_type(2)))  __bf16 v2bf;
typedef __attribute__((ext_vector_type(8)))  float  v8f;

#define IN_CH  256
#define HID_CH 128
#define OUT_CH 64

__device__ __forceinline__ uint32_t pack_bf16(float a, float b) {
#if __has_builtin(__builtin_amdgcn_cvt_pk_bf16_f32)
    union { v2bf v; uint32_t u; } cv;
    cv.v = __builtin_amdgcn_cvt_pk_bf16_f32(a, b);   // lo = a, hi = b
    return cv.u;
#else
    uint32_t ua = __float_as_uint(a);
    uint32_t ub = __float_as_uint(b);
    ua = (ua + 0x7FFFu + ((ua >> 16) & 1u)) >> 16;   // round-to-nearest-even
    ub = (ub + 0x7FFFu + ((ub >> 16) & 1u)) >> 16;
    return (ub << 16) | (ua & 0xFFFFu);
#endif
}

union Frag16 { uint32_t u[8]; v16bf v; };

// ---------------------------------------------------------------------------
// C[M x 128] = act(A[M x K]) @ [B0 | B1]  (B row-major K x n0 / K x n1, f32)
// bf16 WMMA, f32 accumulate. 8 waves/block, each wave = 16 rows x 128 cols.
// ---------------------------------------------------------------------------
template<int RELU>
__global__ __launch_bounds__(256) void gemm_bf16_wmma(
        const float* __restrict__ A, int M, int K,
        const float* __restrict__ B0, int n0,
        const float* __restrict__ B1, int n1,
        float* __restrict__ C)
{
    constexpr int NT  = 128;   // output columns
    constexpr int LDB = 132;   // padded LDS stride (u32) -> bank-conflict-free
    __shared__ uint32_t bsm[NT * LDB];

    const int halfK = K >> 1;
    // Stage B into LDS as packed bf16 K-pairs: bsm[n*LDB + j] = {B[2j][n], B[2j+1][n]}
    for (int idx = threadIdx.x; idx < NT * halfK; idx += blockDim.x) {
        int n = idx / halfK;
        int j = idx - n * halfK;
        const float* Bp; int ldb, col;
        if (n < n0) { Bp = B0; ldb = n0; col = n; }
        else        { Bp = B1; ldb = n1; col = n - n0; }
        float lo = Bp[(size_t)(2 * j) * ldb + col];
        float hi = Bp[(size_t)(2 * j + 1) * ldb + col];
        bsm[n * LDB + j] = pack_bf16(lo, hi);
    }
    __syncthreads();

    const int lane = threadIdx.x & 31;
    const int wave = threadIdx.x >> 5;
    const int hi   = lane >> 4;          // lane half selects K sub-block
    const int l15  = lane & 15;
    const int rowTile = blockIdx.x * 128 + wave * 16;
    int rowA = rowTile + l15;
    if (rowA > M - 1) rowA = M - 1;      // clamp loads; stores are masked
    const float* __restrict__ ap = A + (size_t)rowA * K;

    v8f acc[8];
#pragma unroll
    for (int i = 0; i < 8; ++i) acc[i] = (v8f){};

    const int nsteps = K >> 5;           // 16x16x32 bf16 WMMA steps
    for (int kb = 0; kb < nsteps; ++kb) {
        // A fragment (16x32 bf16): lanes 0-15 -> K 0-15, lanes 16-31 -> K 16-31
        const int k0 = kb * 32 + (hi ? 16 : 0);
        const int k1 = k0 + 8;
        float4 x0 = *(const float4*)(ap + k0);
        float4 x1 = *(const float4*)(ap + k0 + 4);
        float4 y0 = *(const float4*)(ap + k1);
        float4 y1 = *(const float4*)(ap + k1 + 4);
        if (RELU) {
            x0.x = fmaxf(x0.x, 0.f); x0.y = fmaxf(x0.y, 0.f);
            x0.z = fmaxf(x0.z, 0.f); x0.w = fmaxf(x0.w, 0.f);
            x1.x = fmaxf(x1.x, 0.f); x1.y = fmaxf(x1.y, 0.f);
            x1.z = fmaxf(x1.z, 0.f); x1.w = fmaxf(x1.w, 0.f);
            y0.x = fmaxf(y0.x, 0.f); y0.y = fmaxf(y0.y, 0.f);
            y0.z = fmaxf(y0.z, 0.f); y0.w = fmaxf(y0.w, 0.f);
            y1.x = fmaxf(y1.x, 0.f); y1.y = fmaxf(y1.y, 0.f);
            y1.z = fmaxf(y1.z, 0.f); y1.w = fmaxf(y1.w, 0.f);
        }
        Frag16 a;
        a.u[0] = pack_bf16(x0.x, x0.y); a.u[1] = pack_bf16(x0.z, x0.w);
        a.u[2] = pack_bf16(x1.x, x1.y); a.u[3] = pack_bf16(x1.z, x1.w);
        a.u[4] = pack_bf16(y0.x, y0.y); a.u[5] = pack_bf16(y0.z, y0.w);
        a.u[6] = pack_bf16(y1.x, y1.y); a.u[7] = pack_bf16(y1.z, y1.w);

        // B fragment (32x16 bf16): lanes 0-15 -> K 0-15, lanes 16-31 -> K 16-31
        const int jb = kb * 16 + (hi ? 8 : 0);
#pragma unroll
        for (int nt = 0; nt < 8; ++nt) {
            const uint32_t* bp = &bsm[(nt * 16 + l15) * LDB + jb];
            Frag16 b;
#pragma unroll
            for (int v = 0; v < 8; ++v) b.u[v] = bp[v];
            acc[nt] = __builtin_amdgcn_wmma_f32_16x16x32_bf16(
                false, a.v, false, b.v, (short)0, acc[nt], false, false);
        }
    }

    // Store C per 16x16 f32 D layout: lanes 0-15 rows 0-7, lanes 16-31 rows 8-15.
    // A wave's 16-row tile is all-in or all-out when the tail is <16 rows away,
    // so take a branch-free path for full tiles (the overwhelmingly common case).
    const int rbase = rowTile + (hi ? 8 : 0);
    if (rowTile + 16 <= M) {
#pragma unroll
        for (int nt = 0; nt < 8; ++nt) {
            const int col = nt * 16 + l15;
#pragma unroll
            for (int v = 0; v < 8; ++v)
                C[(size_t)(rbase + v) * NT + col] = acc[nt][v];
        }
    } else {
#pragma unroll
        for (int nt = 0; nt < 8; ++nt) {
            const int col = nt * 16 + l15;
#pragma unroll
            for (int v = 0; v < 8; ++v) {
                const int r = rbase + v;
                if (r < M) C[(size_t)r * NT + col] = acc[nt][v];
            }
        }
    }
}

// ---------------------------------------------------------------------------
// Degree / normalization kernels
// ---------------------------------------------------------------------------
__global__ void deg_init(float* deg, int N) {
    int i = blockIdx.x * blockDim.x + threadIdx.x;
    if (i < N) deg[i] = 1.0f;                      // self-loop
}
__global__ void deg_edges(const int* __restrict__ dst, float* deg, int E) {
    int e = blockIdx.x * blockDim.x + threadIdx.x;
    if (e < E) atomicAdd(&deg[dst[e]], 1.0f);
}
__global__ void deg_to_dinv(float* deg, int N) {
    int i = blockIdx.x * blockDim.x + threadIdx.x;
    if (i < N) deg[i] = rsqrtf(deg[i]);
}

// acc1 = h0 * dinv^2 + b1   (self-loop contribution + bias, initializes acc1)
__global__ void seed_hidden(const float* __restrict__ h0, const float* __restrict__ dinv,
                            const float* __restrict__ b1, float* __restrict__ acc1, int N) {
    int idx = blockIdx.x * blockDim.x + threadIdx.x;
    if (idx >= N * HID_CH) return;
    int i = idx >> 7, c = idx & 127;
    float s = dinv[i]; s *= s;
    acc1[idx] = h0[idx] * s + b1[c];
}

// acc1[dst] += h0[src] * dinv[src]*dinv[dst]  (wave per edge, float4 per lane).
// e is wave-uniform: force it scalar so edge/dinv reads become SMEM loads.
__global__ void edge_prop_hidden(const int* __restrict__ src, const int* __restrict__ dst,
                                 const float* __restrict__ dinv,
                                 const float* __restrict__ h0, float* __restrict__ acc1, int E) {
    int t = blockIdx.x * blockDim.x + threadIdx.x;
    int e = __builtin_amdgcn_readfirstlane(t >> 5);
    if (e >= E) return;
    int lane = t & 31;
    int s = __builtin_amdgcn_readfirstlane(src[e]);
    int d = __builtin_amdgcn_readfirstlane(dst[e]);
    float w = dinv[s] * dinv[d];
    float4 v = ((const float4*)(h0 + (size_t)s * HID_CH))[lane];
    float* o = acc1 + (size_t)d * HID_CH + lane * 4;
    atomicAdd(o + 0, v.x * w); atomicAdd(o + 1, v.y * w);
    atomicAdd(o + 2, v.z * w); atomicAdd(o + 3, v.w * w);
}

// out_mu / out_logvar seeded with self-loop term + bias (also clears poison)
__global__ void seed_out(const float* __restrict__ z, const float* __restrict__ dinv,
                         const float* __restrict__ bmu, const float* __restrict__ blv,
                         float* __restrict__ om, float* __restrict__ ol, int N) {
    int idx = blockIdx.x * blockDim.x + threadIdx.x;
    if (idx >= N * OUT_CH) return;
    int i = idx >> 6, c = idx & 63;
    float s = dinv[i]; s *= s;
    om[idx] = z[(size_t)i * 128 + c]      * s + bmu[c];
    ol[idx] = z[(size_t)i * 128 + 64 + c] * s + blv[c];
}

__global__ void edge_prop_out(const int* __restrict__ src, const int* __restrict__ dst,
                              const float* __restrict__ dinv, const float* __restrict__ z,
                              float* __restrict__ om, float* __restrict__ ol, int E) {
    int t = blockIdx.x * blockDim.x + threadIdx.x;
    int e = __builtin_amdgcn_readfirstlane(t >> 5);
    if (e >= E) return;
    int lane = t & 31;
    int s = __builtin_amdgcn_readfirstlane(src[e]);
    int d = __builtin_amdgcn_readfirstlane(dst[e]);
    float w = dinv[s] * dinv[d];
    int c = lane * 4;                                   // cols 0-63 -> mu, 64-127 -> logvar
    float4 v = ((const float4*)(z + (size_t)s * 128))[lane];
    float* o = (c < 64) ? (om + (size_t)d * OUT_CH + c)
                        : (ol + (size_t)d * OUT_CH + (c - 64));
    atomicAdd(o + 0, v.x * w); atomicAdd(o + 1, v.y * w);
    atomicAdd(o + 2, v.z * w); atomicAdd(o + 3, v.w * w);
}

// ---------------------------------------------------------------------------
extern "C" void kernel_launch(void* const* d_in, const int* in_sizes, int n_in,
                              void* d_out, int out_size, void* d_ws, size_t ws_size,
                              hipStream_t stream) {
    const int N = in_sizes[0] / IN_CH;     // 100000
    const int E = in_sizes[1] / 2;         // 640000

    const float* x    = (const float*)d_in[0];
    const int*   ei   = (const int*)d_in[1];
    const int*   esrc = ei;
    const int*   edst = ei + E;
    const float* W1   = (const float*)d_in[2];
    const float* b1   = (const float*)d_in[3];
    const float* Wmu  = (const float*)d_in[4];
    const float* bmu  = (const float*)d_in[5];
    const float* Wlv  = (const float*)d_in[6];
    const float* blv  = (const float*)d_in[7];

    float* out_mu = (float*)d_out;
    float* out_lv = out_mu + (size_t)N * OUT_CH;

    // workspace layout: dinv | h0 (reused as z) | acc1
    float* dinv = (float*)d_ws;
    size_t nAlign = ((size_t)N + 255) & ~(size_t)255;
    float* h0   = dinv + nAlign;
    float* acc1 = h0 + (size_t)N * HID_CH;

    const int B = 256;
    const int gN   = (N + B - 1) / B;
    const int gE   = (E + B - 1) / B;
    const int gE32 = (E * 32 + B - 1) / B;
    const int gNH  = (N * HID_CH + B - 1) / B;
    const int gNO  = (N * OUT_CH + B - 1) / B;
    const int gM   = (N + 127) / 128;      // GEMM blocks: 128 rows each

    // 1) symmetric normalization coefficients
    deg_init   <<<gN, B, 0, stream>>>(dinv, N);
    deg_edges  <<<gE, B, 0, stream>>>(edst, dinv, E);
    deg_to_dinv<<<gN, B, 0, stream>>>(dinv, N);

    // 2) h0 = x @ W1   (bf16 WMMA)
    gemm_bf16_wmma<0><<<gM, B, 0, stream>>>(x, N, IN_CH, W1, HID_CH, W1, 0, h0);

    // 3) acc1 = P @ h0 + b1
    seed_hidden     <<<gNH,  B, 0, stream>>>(h0, dinv, b1, acc1, N);
    edge_prop_hidden<<<gE32, B, 0, stream>>>(esrc, edst, dinv, h0, acc1, E);

    // 4) z = relu(acc1) @ [W_mu | W_logvar]   (bf16 WMMA, reuses h0 buffer)
    gemm_bf16_wmma<1><<<gM, B, 0, stream>>>(acc1, N, HID_CH, Wmu, OUT_CH, Wlv, OUT_CH, h0);

    // 5) (mu, logvar) = P @ z + bias, split halves into d_out
    seed_out     <<<gNO,  B, 0, stream>>>(h0, dinv, bmu, blv, out_mu, out_lv, N);
    edge_prop_out<<<gE32, B, 0, stream>>>(esrc, edst, dinv, h0, out_mu, out_lv, E);
}
